// GUNet1L_uw_54485955117464
// MI455X (gfx1250) — compile-verified, matrix operand order checked by reading, not run on hardware
//
#include <hip/hip_runtime.h>

// ---------------------------------------------------------------------------
// GraphUNet (dense) for MI455X / gfx1250.
// Heavy ops (A1@A1 augment ~137 GFLOP, all GCN matmuls) run through
// v_wmma_f32_16x16x32_bf16 (fp32 accumulate). GEMM operands are pre-converted
// to bf16 in global memory (halves operand traffic for the traffic-bound
// augment GEMM). A-tile staging uses CDNA5 async global->LDS copies
// (GLOBAL_LOAD_ASYNC_TO_LDS_B128, ASYNCcnt) to skip the VGPR round-trip.
// ---------------------------------------------------------------------------

typedef __attribute__((ext_vector_type(16))) __bf16          v16bf;
typedef __attribute__((ext_vector_type(8)))  float           v8f;
typedef __attribute__((ext_vector_type(8)))  unsigned short  us8;
typedef __attribute__((ext_vector_type(4)))  unsigned short  us4;

#define TM 128
#define TN 64
#define KT 64                  // two 16x16x32 WMMA K-steps per staged tile
#define LDS_STRIDE (KT + 8)    // 72 ushorts = 144B: rows 16B-aligned, banks spread
#define C_FEAT 256

__device__ __forceinline__ unsigned short f2bf(float f) {
    // round-to-nearest-even fp32 -> bf16 truncation
    unsigned int x = __float_as_uint(f);
    unsigned int r = x + 0x7FFFu + ((x >> 16) & 1u);
    return (unsigned short)(r >> 16);
}

union FragBF { v16bf v; unsigned short u[16]; };

// ---------------------------------------------------------------------------
// Vectorized fp32 -> bf16 conversion (element count multiple of 4)
// ---------------------------------------------------------------------------
__global__ void f32_to_bf16(const float* __restrict__ src,
                            unsigned short* __restrict__ dst, long long n4) {
    long long t = (long long)blockIdx.x * 256 + threadIdx.x;
    if (t < n4) {
        const float4 v = ((const float4*)src)[t];
        us4 o;
        o[0] = f2bf(v.x); o[1] = f2bf(v.y); o[2] = f2bf(v.z); o[3] = f2bf(v.w);
        ((us4*)dst)[t] = o;
    }
}

// ---------------------------------------------------------------------------
// Tiled bf16 WMMA GEMM: C = A(MxK,row-major,lda) @ B(KxN,row-major,ldb)
// A,B are bf16 bit-patterns (ushort). Block: 256 threads = 8 waves (4x2),
// macro-tile 128x64, wave tile 32x32, 8 v_wmma per wave per staged K-tile.
// ---------------------------------------------------------------------------
__global__ __launch_bounds__(256)
void gemm_bf16_wmma(const unsigned short* __restrict__ A, int lda,
                    const unsigned short* __restrict__ B, int ldb,
                    float* __restrict__ C, int ldc,
                    int M, int N, int K)
{
    __shared__ unsigned short As[TM][LDS_STRIDE];  // A tile 128x64 (bf16)
    __shared__ unsigned short Bs[TN][LDS_STRIDE];  // B^T tile 64x64 (bf16)

    const int wave = threadIdx.x >> 5;
    const int lane = threadIdx.x & 31;
    const int mw = wave >> 1;     // 0..3 -> M offset 32*mw
    const int nw = wave & 1;      // 0..1 -> N offset 32*nw
    const int blockM = blockIdx.y * TM;
    const int blockN = blockIdx.x * TN;

    // Uniform per-block fast-path test: every GEMM here except the final
    // 4096x16 linear layer has M,N multiples of the macro-tile and K%64==0.
    const bool fullMN = (blockM + TM <= M) && (blockN + TN <= N);

    v8f acc[2][2] = {};           // [mTile][nTile]

    const int kTiles = (K + KT - 1) / KT;
    for (int kt = 0; kt < kTiles; ++kt) {
        const int k0 = kt * KT;
        if (fullMN && (k0 + KT <= K)) {
            // ---- fast staging ----
            // A tile: contiguous 16B/lane copies -> async global->LDS DMA,
            // no VGPR round-trip, tracked on ASYNCcnt.
            #pragma unroll
            for (int it = 0; it < (TM * KT / 8) / 256; ++it) {
                int t = it * 256 + threadIdx.x;
                int r = t >> 3;             // KT/8 = 8 octets per row
                int ko = (t & 7) * 8;
                const unsigned short* gp =
                    A + (size_t)(blockM + r) * lda + (k0 + ko);
                unsigned ldsAddr =
                    (unsigned)(unsigned long long)(uintptr_t)&As[r][ko];
                asm volatile("global_load_async_to_lds_b128 %0, %1, off"
                             :: "v"(ldsAddr),
                                "v"((unsigned long long)(uintptr_t)gp)
                             : "memory");
            }
            // B tile transposed: clause the global loads, then 16-bit stores
            us8 bv[((TN / 8) * KT) / 256];
            #pragma unroll
            for (int it = 0; it < ((TN / 8) * KT) / 256; ++it) {
                int t = it * 256 + threadIdx.x;
                int kk = t >> 3;            // TN/8 = 8 octets per k-row
                int nq = (t & 7) * 8;
                bv[it] = *(const us8*)
                    (B + (size_t)(k0 + kk) * ldb + (blockN + nq));
            }
            #pragma unroll
            for (int it = 0; it < ((TN / 8) * KT) / 256; ++it) {
                int t = it * 256 + threadIdx.x;
                int kk = t >> 3;
                int nq = (t & 7) * 8;
                #pragma unroll
                for (int j = 0; j < 8; ++j) Bs[nq + j][kk] = bv[it][j];
            }
            // collect the async A-tile copies before publishing the barrier
            asm volatile("s_wait_asynccnt 0x0" ::: "memory");
        } else {
            // ---- guarded fallback (final 256x16 linear layer only)
            for (int t = threadIdx.x; t < TM * KT; t += 256) {
                int r = t >> 6, c = t & 63;
                int gr = blockM + r, gc = k0 + c;
                As[r][c] = (gr < M && gc < K) ?
                    A[(size_t)gr * lda + gc] : (unsigned short)0;
            }
            for (int t = threadIdx.x; t < TN * KT; t += 256) {
                int nn = t & 63, kk = t >> 6;
                int gk = k0 + kk, gn = blockN + nn;
                Bs[nn][kk] = (gk < K && gn < N) ?
                    B[(size_t)gk * ldb + gn] : (unsigned short)0;
            }
        }
        __syncthreads();

        // Fragment layout (ISA 7.12.2, 16-bit A 16x32): lane = (kh<<4)|row,
        // lane holds K = kh*8 + {0..7} and kh*8 + 16 + {0..7} (two 16B runs).
        const int kh = lane >> 4;
        const int lr = lane & 15;

        #pragma unroll
        for (int kk = 0; kk < 2; ++kk) {    // two 16x16x32 K steps per tile
            FragBF aF[2], bF[2];
            #pragma unroll
            for (int mt = 0; mt < 2; ++mt) {
                const unsigned short* ar = &As[mw * 32 + mt * 16 + lr][kk * 32];
                #pragma unroll
                for (int t = 0; t < 8; ++t) {
                    aF[mt].u[t]     = ar[kh * 8 + t];
                    aF[mt].u[8 + t] = ar[kh * 8 + 16 + t];
                }
            }
            #pragma unroll
            for (int nt = 0; nt < 2; ++nt) {
                const unsigned short* br = &Bs[nw * 32 + nt * 16 + lr][kk * 32];
                #pragma unroll
                for (int t = 0; t < 8; ++t) {
                    bF[nt].u[t]     = br[kh * 8 + t];
                    bF[nt].u[8 + t] = br[kh * 8 + 16 + t];
                }
            }
            #pragma unroll
            for (int mt = 0; mt < 2; ++mt)
                #pragma unroll
                for (int nt = 0; nt < 2; ++nt)
                    acc[mt][nt] = __builtin_amdgcn_wmma_f32_16x16x32_bf16(
                        false, aF[mt].v, false, bF[nt].v,
                        (short)0, acc[mt][nt], false, false);
        }
        __syncthreads();
    }

    // C/D layout: lanes 0-15: M=vgpr, N=lane; lanes 16-31: M=vgpr+8, N=lane-16
    const int cn = lane & 15;
    const int mhalf = (lane >> 4) * 8;
    #pragma unroll
    for (int mt = 0; mt < 2; ++mt)
        #pragma unroll
        for (int nt = 0; nt < 2; ++nt) {
            int gn = blockN + nw * 32 + nt * 16 + cn;
            if (gn >= N) continue;
            int gm0 = blockM + mw * 32 + mt * 16 + mhalf;
            #pragma unroll
            for (int v = 0; v < 8; ++v) {
                int gm = gm0 + v;
                if (gm < M) C[(size_t)gm * ldc + gn] = acc[mt][nt][v];
            }
        }
}

// ---------------------------------------------------------------------------
// Small kernels (all bandwidth-trivial)
// ---------------------------------------------------------------------------
__global__ void build_adj(const int* __restrict__ ei, int E, float* A, int n) {
    int e = blockIdx.x * 256 + threadIdx.x;
    if (e < E) {
        int s = ei[e], d = ei[E + e];
        atomicAdd(&A[(size_t)d * n + s], 1.0f);
    }
}

// per-row: deg of (A + diag(d0>0?0:2)), dis = rsqrt(deg), dxtra = selfloop fix
__global__ void gcn_prep(const float* __restrict__ A, int n, float* dis, float* dxtra) {
    int i = blockIdx.x;
    __shared__ float red[256];
    float s = 0.f;
    for (int j = threadIdx.x; j < n; j += 256) s += A[(size_t)i * n + j];
    red[threadIdx.x] = s; __syncthreads();
    for (int w = 128; w > 0; w >>= 1) {
        if (threadIdx.x < w) red[threadIdx.x] += red[threadIdx.x + w];
        __syncthreads();
    }
    if (threadIdx.x == 0) {
        float d0 = A[(size_t)i * n + i];
        float ex = (d0 > 0.f) ? 0.f : 2.f;
        dis[i] = rsqrtf(red[0] + ex);
        dxtra[i] = ex;
    }
}

__global__ void scale_rows(float* z, const float* __restrict__ dis, int n) {
    long long t = (long long)blockIdx.x * 256 + threadIdx.x;
    if (t < (long long)n * C_FEAT) z[t] *= dis[t >> 8];
}

// h = dis_i*(y + dxtra_i*z) + b, optional relu
__global__ void gcn_epilogue(const float* __restrict__ y, const float* __restrict__ z,
                             const float* __restrict__ dis, const float* __restrict__ dxtra,
                             const float* __restrict__ b, int relu, float* h, int n) {
    long long t = (long long)blockIdx.x * 256 + threadIdx.x;
    if (t < (long long)n * C_FEAT) {
        int i = (int)(t >> 8), c = (int)(t & 255);
        float v = dis[i] * (y[t] + dxtra[i] * z[t]) + b[c];
        h[t] = relu ? fmaxf(v, 0.f) : v;
    }
}

__global__ void make_A1(const float* __restrict__ A, int n, float* A1) {
    long long t = (long long)blockIdx.x * 256 + threadIdx.x;
    long long nn = (long long)n * n;
    if (t < nn) {
        long long r = t / n, c = t - r * n;
        A1[t] = (r == c) ? 1.0f : A[t];
    }
}

__global__ void pool_score(const float* __restrict__ h, const float* __restrict__ w,
                           int n, float* score) {
    int i = blockIdx.x;
    __shared__ float rd[256], rn[256];
    float wc = w[threadIdx.x];
    float hv = h[(size_t)i * C_FEAT + threadIdx.x];
    rd[threadIdx.x] = hv * wc; rn[threadIdx.x] = wc * wc; __syncthreads();
    for (int s = 128; s > 0; s >>= 1) {
        if (threadIdx.x < s) { rd[threadIdx.x] += rd[threadIdx.x + s];
                               rn[threadIdx.x] += rn[threadIdx.x + s]; }
        __syncthreads();
    }
    if (threadIdx.x == 0) score[i] = tanhf(rd[0] / sqrtf(rn[0]));
}

// Single-block bitonic sort (desc by value, tie -> lower index first, matching
// jax top_k). n <= 4096 and pow2 here; pads with -inf.
__global__ void topk_sort(const float* __restrict__ score, int n, int k,
                          int* perm, float* vals) {
    __shared__ float sv[4096];
    __shared__ int   si[4096];
    const int T = 512;
    for (int t = threadIdx.x; t < 4096; t += T) {
        sv[t] = (t < n) ? score[t] : -3.0e38f;
        si[t] = t;
    }
    __syncthreads();
    for (int size = 2; size <= 4096; size <<= 1) {
        for (int stride = size >> 1; stride > 0; stride >>= 1) {
            for (int t = threadIdx.x; t < 2048; t += T) {
                int pos = 2 * t - (t & (stride - 1));
                int a = pos, b = pos + stride;
                bool descBlock = ((a & size) == 0);
                float va = sv[a], vb = sv[b];
                int ia = si[a], ib = si[b];
                bool aFirst = (va > vb) || (va == vb && ia < ib);
                bool doSwap = descBlock ? (!aFirst) : aFirst;
                if (doSwap) { sv[a] = vb; sv[b] = va; si[a] = ib; si[b] = ia; }
            }
            __syncthreads();
        }
    }
    for (int t = threadIdx.x; t < k; t += T) { perm[t] = si[t]; vals[t] = sv[t]; }
}

__global__ void gather_gate(const float* __restrict__ h, const int* __restrict__ perm,
                            const float* __restrict__ vals, int k, float* g) {
    long long t = (long long)blockIdx.x * 256 + threadIdx.x;
    if (t < (long long)k * C_FEAT) {
        int r = (int)(t >> 8), c = (int)(t & 255);
        g[t] = h[(size_t)perm[r] * C_FEAT + c] * vals[r];
    }
}

// dst[r][c] = A2[perm[r]][perm[c]]; diagonal of augmented product is 0 by def.
__global__ void gather_adj(const float* __restrict__ A2, int n,
                           const int* __restrict__ perm, int k, float* dst) {
    long long t = (long long)blockIdx.x * 256 + threadIdx.x;
    if (t < (long long)k * k) {
        int r = (int)(t / k), c = (int)(t - (long long)r * k);
        dst[t] = (r == c) ? 0.0f : A2[(size_t)perm[r] * n + perm[c]];
    }
}

__global__ void scatter_add(float* dst, const float* __restrict__ h,
                            const int* __restrict__ perm, int k) {
    long long t = (long long)blockIdx.x * 256 + threadIdx.x;
    if (t < (long long)k * C_FEAT) {
        int r = (int)(t >> 8), c = (int)(t & 255);
        dst[(size_t)perm[r] * C_FEAT + c] += h[t];
    }
}

__global__ void bn_stats(const float* __restrict__ h, int n, float* mean, float* var) {
    int c = blockIdx.x;
    __shared__ float s1[256], s2[256];
    float a = 0.f, b = 0.f;
    for (int i = threadIdx.x; i < n; i += 256) {
        float v = h[(size_t)i * C_FEAT + c];
        a += v; b += v * v;
    }
    s1[threadIdx.x] = a; s2[threadIdx.x] = b; __syncthreads();
    for (int s = 128; s > 0; s >>= 1) {
        if (threadIdx.x < s) { s1[threadIdx.x] += s1[threadIdx.x + s];
                               s2[threadIdx.x] += s2[threadIdx.x + s]; }
        __syncthreads();
    }
    if (threadIdx.x == 0) {
        float m = s1[0] / n;
        mean[c] = m;
        var[c] = s2[0] / n - m * m;   // biased variance, as reference
    }
}

__global__ void bn_apply(const float* __restrict__ h, const float* __restrict__ mean,
                         const float* __restrict__ var, const float* __restrict__ gamma,
                         const float* __restrict__ beta, float* g, int n) {
    long long t = (long long)blockIdx.x * 256 + threadIdx.x;
    if (t < (long long)n * C_FEAT) {
        int c = (int)(t & 255);
        g[t] = (h[t] - mean[c]) * rsqrtf(var[c] + 1e-5f) * gamma[c] + beta[c];
    }
}

__global__ void add_bias16(const float* __restrict__ y, const float* __restrict__ b,
                           float* out, int n) {
    int t = blockIdx.x * 256 + threadIdx.x;
    if (t < n * 16) out[t] = y[t] + b[t & 15];
}

// ---------------------------------------------------------------------------
// Host orchestration
// ---------------------------------------------------------------------------
static inline void launch_gemm(const unsigned short* A, int lda,
                               const unsigned short* B, int ldb,
                               float* C, int ldc, int M, int N, int K,
                               hipStream_t s) {
    dim3 grid((N + TN - 1) / TN, (M + TM - 1) / TM);
    gemm_bf16_wmma<<<grid, dim3(256), 0, s>>>(A, lda, B, ldb, C, ldc, M, N, K);
}

static inline void to_bf16(const float* src, unsigned short* dst, long long n,
                           hipStream_t s) {
    long long n4 = n / 4;
    f32_to_bf16<<<(unsigned)((n4 + 255) / 256), 256, 0, s>>>(src, dst, n4);
}

extern "C" void kernel_launch(void* const* d_in, const int* in_sizes, int n_in,
                              void* d_out, int out_size, void* d_ws, size_t ws_size,
                              hipStream_t stream) {
    (void)n_in; (void)out_size; (void)ws_size;
    const int C = C_FEAT, OUT = 16;
    const int ns[4] = {4096, 2048, 1024, 512};

    const float* x        = (const float*)d_in[0];
    const int*   edge_idx = (const int*)d_in[1];
    const float* down_W   = (const float*)d_in[2];
    const float* down_b   = (const float*)d_in[3];
    const float* pool_w   = (const float*)d_in[4];
    const float* up_W     = (const float*)d_in[5];
    const float* up_b     = (const float*)d_in[6];
    const float* bn_gamma = (const float*)d_in[7];
    const float* bn_beta  = (const float*)d_in[8];
    const float* lin_W    = (const float*)d_in[9];
    const float* lin_b    = (const float*)d_in[10];
    float* out = (float*)d_out;
    const int E = in_sizes[1] / 2;

    // workspace carve-out (float granularity)
    float* p = (float*)d_ws;
    auto alloc = [&](size_t nfl) { float* r = p; p += nfl; return r; };
    float* As0  = alloc((size_t)4096 * 4096);
    float* As1  = alloc((size_t)2048 * 2048);
    float* As2  = alloc((size_t)1024 * 1024);
    float* A3   = alloc((size_t)512 * 512);
    float* Aug1 = alloc((size_t)4096 * 4096);
    float* Aug2 = alloc((size_t)4096 * 4096);
    float* h    = alloc((size_t)4096 * C);
    float* g    = alloc((size_t)4096 * C);
    float* t1   = alloc((size_t)4096 * C);
    float* t2   = alloc((size_t)4096 * C);
    float* xs0  = alloc((size_t)4096 * C);
    float* xs1  = alloc((size_t)2048 * C);
    float* xs2  = alloc((size_t)1024 * C);
    float* dis  = alloc(4096);
    float* dx   = alloc(4096);
    float* score= alloc(4096);
    float* vals = alloc(4096);
    float* mean = alloc(256);
    float* var  = alloc(256);
    int* perm0 = (int*)alloc(2048);
    int* perm1 = (int*)alloc(1024);
    int* perm2 = (int*)alloc(512);
    // bf16 operand mirrors (ushort, counted in float slots = 2 ushorts each)
    unsigned short* bfBig = (unsigned short*)alloc((size_t)4096 * 4096 / 2); // n x n
    unsigned short* bfS1  = (unsigned short*)alloc((size_t)4096 * C / 2);    // n x C
    unsigned short* bfS2  = (unsigned short*)alloc((size_t)C * C / 2);       // C x C

    float* AsArr[3]   = {As0, As1, As2};
    int*   permArr[3] = {perm0, perm1, perm2};
    float* xsArr[3]   = {xs0, xs1, xs2};

    auto gcn = [&](const float* A, int n, const float* X, const float* W,
                   const float* b, int relu, float* outH) {
        gcn_prep<<<n, 256, 0, stream>>>(A, n, dis, dx);
        to_bf16(X, bfS1, (long long)n * C, stream);
        to_bf16(W, bfS2, (long long)C * C, stream);
        launch_gemm(bfS1, C, bfS2, C, t1, C, n, C, C, stream);    // XW
        long long e = (long long)n * C;
        unsigned eb = (unsigned)((e + 255) / 256);
        scale_rows<<<eb, 256, 0, stream>>>(t1, dis, n);           // z' = dis ⊙ XW
        to_bf16(A, bfBig, (long long)n * n, stream);
        to_bf16(t1, bfS1, (long long)n * C, stream);
        launch_gemm(bfBig, n, bfS1, C, t2, C, n, C, n, stream);   // y = A @ z'
        gcn_epilogue<<<eb, 256, 0, stream>>>(t2, t1, dis, dx, b, relu, outH, n);
    };

    // Build dense adjacency
    hipMemsetAsync(As0, 0, (size_t)4096 * 4096 * sizeof(float), stream);
    build_adj<<<(E + 255) / 256, 256, 0, stream>>>(edge_idx, E, As0, 4096);

    // First GCN + save residual
    gcn(As0, 4096, x, down_W, down_b, 1, h);
    hipMemcpyAsync(xs0, h, (size_t)4096 * C * sizeof(float),
                   hipMemcpyDeviceToDevice, stream);

    // Down path
    const float* curA = As0;
    for (int i = 0; i < 3; ++i) {
        int n = ns[i], k = ns[i + 1];
        long long nn = (long long)n * n;
        make_A1<<<(unsigned)((nn + 255) / 256), 256, 0, stream>>>(curA, n, Aug1);
        to_bf16(Aug1, bfBig, nn, stream);                          // one convert,
        launch_gemm(bfBig, n, bfBig, n, Aug2, n, n, n, n, stream); // both operands
        pool_score<<<n, 256, 0, stream>>>(h, pool_w + (size_t)i * C, n, score);
        topk_sort<<<1, 512, 0, stream>>>(score, n, k, permArr[i], vals);
        long long kc = (long long)k * C;
        gather_gate<<<(unsigned)((kc + 255) / 256), 256, 0, stream>>>(
            h, permArr[i], vals, k, g);
        float* nextA = (i < 2) ? AsArr[i + 1] : A3;
        long long kk = (long long)k * k;
        gather_adj<<<(unsigned)((kk + 255) / 256), 256, 0, stream>>>(
            Aug2, n, permArr[i], k, nextA);
        gcn(nextA, k, g, down_W + (size_t)(i + 1) * C * C,
            down_b + (size_t)(i + 1) * C, 1, h);
        if (i < 2)
            hipMemcpyAsync(xsArr[i + 1], h, (size_t)k * C * sizeof(float),
                           hipMemcpyDeviceToDevice, stream);
        curA = nextA;
    }

    // Up path (last GCN's missing inner relu is replaced by the outer relu)
    for (int i = 0; i < 3; ++i) {
        int j = 2 - i;
        int nj = ns[j], k = ns[j + 1];
        hipMemcpyAsync(g, xsArr[j], (size_t)nj * C * sizeof(float),
                       hipMemcpyDeviceToDevice, stream);
        long long kc = (long long)k * C;
        scatter_add<<<(unsigned)((kc + 255) / 256), 256, 0, stream>>>(
            g, h, permArr[j], k);
        gcn(AsArr[j], nj, g, up_W + (size_t)i * C * C, up_b + (size_t)i * C, 1, h);
    }

    // BatchNorm (training-mode batch stats) + final linear
    bn_stats<<<C, 256, 0, stream>>>(h, 4096, mean, var);
    long long e0 = (long long)4096 * C;
    bn_apply<<<(unsigned)((e0 + 255) / 256), 256, 0, stream>>>(
        h, mean, var, bn_gamma, bn_beta, g, 4096);
    to_bf16(g, bfS1, e0, stream);
    to_bf16(lin_W, bfS2, (long long)C * OUT, stream);
    launch_gemm(bfS1, C, bfS2, OUT, t1, OUT, 4096, OUT, C, stream);
    add_bias16<<<(4096 * 16 + 255) / 256, 256, 0, stream>>>(t1, lin_b, out, 4096);
}
